// MultiHeadAttention_21758304322196
// MI455X (gfx1250) — matrix operand order repaired and checked by reading, not compiled, not run on hardware
//
#include <hip/hip_runtime.h>
#include <hip/hip_bf16.h>

#define BB 4
#define SS 1024
#define DD 1024
#define HH 16
#define CHN 64

typedef __attribute__((ext_vector_type(16))) __bf16 v16bf;
typedef __attribute__((ext_vector_type(8)))  __bf16 v8bf;
typedef __attribute__((ext_vector_type(4)))  __bf16 v4bf;
typedef __attribute__((ext_vector_type(8)))  float  v8f;
typedef __attribute__((ext_vector_type(4)))  float  v4f;

// ---------------------------------------------------------------------------
// fp32 -> bf16 conversion (x4 vectorized; sizes are multiples of 1024)
// ---------------------------------------------------------------------------
__global__ __launch_bounds__(256) void cvt_bf16_kernel(const float* __restrict__ src,
                                                       __bf16* __restrict__ dst, int n4) {
    int i = blockIdx.x * 256 + threadIdx.x;
    if (i < n4) {
        v4f x = *(const v4f*)(src + 4 * (size_t)i);
        v4bf y;
        #pragma unroll
        for (int e = 0; e < 4; ++e) y[e] = (__bf16)x[e];
        *(v4bf*)(dst + 4 * (size_t)i) = y;
    }
}

// ---------------------------------------------------------------------------
// Register-blocked GEMM core: each wave computes a 32x32 output block
// (2x2 tiles of 16x16) of  A[M,K](bf16,rm) @ W[N,K](bf16,rm)^T.
// K contracted in chunks of 32 via v_wmma_f32_16x16x32_bf16.
// A-frag: lane holds row m+(lane&15); lanes<16 carry K 0..15,
//         lanes>=16 carry K 16..31 (contiguous, same order as B).
// B-frag: lane holds column n+(lane&15) = W row (contiguous K);
//         lanes<16 K 0..15, lanes>=16 K 16..31 (contiguous order, same as A).
// 4 WMMAs per 8 b128 loads -> 2x the math density of unblocked tiles.
// ---------------------------------------------------------------------------
__device__ __forceinline__ void gemm_tile_2x2(const __bf16* __restrict__ A,
                                              const __bf16* __restrict__ W,
                                              int m0, int n0, int Kdim, int lane,
                                              v8f* acc) {
    const int hi = lane >> 4;
    const __bf16* a0p = A + (size_t)(m0 + (lane & 15)) * Kdim + (hi << 4);
    const __bf16* a1p = a0p + (size_t)16 * Kdim;
    const __bf16* b0p = W + (size_t)(n0 + (lane & 15)) * Kdim + (hi << 4);
    const __bf16* b1p = b0p + (size_t)16 * Kdim;
    for (int kb = 0; kb < Kdim; kb += 32) {
        v8bf t0 = *(const v8bf*)(a0p + kb);
        v8bf t1 = *(const v8bf*)(a0p + kb + 8);
        v8bf t2 = *(const v8bf*)(a1p + kb);
        v8bf t3 = *(const v8bf*)(a1p + kb + 8);
        v16bf a0, a1;
        #pragma unroll
        for (int i = 0; i < 8; ++i) {
            a0[i] = t0[i]; a0[i + 8] = t1[i];
            a1[i] = t2[i]; a1[i + 8] = t3[i];
        }
        v16bf b0 = *(const v16bf*)(b0p + kb);
        v16bf b1 = *(const v16bf*)(b1p + kb);
        acc[0] = __builtin_amdgcn_wmma_f32_16x16x32_bf16(false, a0, false, b0,
                                                         (short)0, acc[0], false, false);
        acc[1] = __builtin_amdgcn_wmma_f32_16x16x32_bf16(false, a0, false, b1,
                                                         (short)0, acc[1], false, false);
        acc[2] = __builtin_amdgcn_wmma_f32_16x16x32_bf16(false, a1, false, b0,
                                                         (short)0, acc[2], false, false);
        acc[3] = __builtin_amdgcn_wmma_f32_16x16x32_bf16(false, a1, false, b1,
                                                         (short)0, acc[3], false, false);
    }
}

// ---------------------------------------------------------------------------
// Q/K/V projections: y = x @ W^T + b, Q additionally scaled by CH^-0.5.
// Output scattered to per-head layout [B,H,S,CH] in bf16.
// blockIdx.z selects Q(0)/K(1)/V(2). 8 waves/block, 32x32 block per wave.
// ---------------------------------------------------------------------------
__global__ __launch_bounds__(256) void proj_qkv_kernel(
    const __bf16* __restrict__ xq, const __bf16* __restrict__ xkv,
    const __bf16* __restrict__ Wqb, const __bf16* __restrict__ Wkb,
    const __bf16* __restrict__ Wvb,
    const float* __restrict__ bq, const float* __restrict__ bk,
    const float* __restrict__ bv,
    __bf16* __restrict__ Qh, __bf16* __restrict__ Kh, __bf16* __restrict__ Vh) {
    const int lane = threadIdx.x & 31;
    const int wave = threadIdx.x >> 5;
    const int tile = blockIdx.x * 8 + wave;   // 4096 wave-blocks = 128 x 32
    const int m0 = (tile >> 5) * 32;
    const int n0 = (tile & 31) * 32;

    const __bf16* A; const __bf16* W; const float* bias; __bf16* dst; float scale;
    if (blockIdx.z == 0)      { A = xq;  W = Wqb; bias = bq; dst = Qh; scale = 0.125f; }
    else if (blockIdx.z == 1) { A = xkv; W = Wkb; bias = bk; dst = Kh; scale = 1.0f; }
    else                      { A = xkv; W = Wvb; bias = bv; dst = Vh; scale = 1.0f; }

    v8f acc[4] = { {}, {}, {}, {} };
    gemm_tile_2x2(A, W, m0, n0, DD, lane, acc);

    #pragma unroll
    for (int mt = 0; mt < 2; ++mt) {
        #pragma unroll
        for (int nt = 0; nt < 2; ++nt) {
            const v8f& a = acc[mt * 2 + nt];
            const int n = n0 + nt * 16 + (lane & 15);
            const float bi = bias[n];
            const int h = n >> 6, c = n & (CHN - 1);
            #pragma unroll
            for (int r = 0; r < 8; ++r) {
                int m = m0 + mt * 16 + r + ((lane >> 4) << 3);  // row in [0, B*S)
                int bidx = m >> 10;                             // / S
                int s = m & (SS - 1);
                float val = (a[r] + bi) * scale;
                dst[((((size_t)bidx * HH + h) * SS) + s) * CHN + c] = (__bf16)val;
            }
        }
    }
}

// ---------------------------------------------------------------------------
// Flash attention (causal + key padding mask), one block per (b,h,128 rows).
// 8 waves, each owns a 16-row query tile. 32-key chunks staged in LDS
// (V transposed). Online softmax; P transposed C->A layout through LDS.
// Output merged heads -> [B,S,D] bf16.
// ---------------------------------------------------------------------------
__global__ __launch_bounds__(256) void attn_kernel(
    const __bf16* __restrict__ Qh, const __bf16* __restrict__ Kh,
    const __bf16* __restrict__ Vh, const bool* __restrict__ pad,
    __bf16* __restrict__ Obf) {
    __shared__ __attribute__((aligned(32))) __bf16 ldsK[32 * CHN];   // [key][ch]
    __shared__ __attribute__((aligned(32))) __bf16 ldsVt[CHN * 32];  // [ch][key]
    __shared__ __attribute__((aligned(32))) float  ldsP[8][16 * 32]; // per-wave P

    const int lane = threadIdx.x & 31;
    const int wave = threadIdx.x >> 5;
    const int hi   = lane >> 4;
    const int bh = blockIdx.y;
    const int b = bh >> 4;
    const int h = bh & (HH - 1);
    const int q0b = blockIdx.x * 128;
    const int qt = q0b + wave * 16;

    const __bf16* Qp = Qh + (size_t)bh * SS * CHN;
    const __bf16* Kp = Kh + (size_t)bh * SS * CHN;
    const __bf16* Vp = Vh + (size_t)bh * SS * CHN;

    // Preload Q A-fragments (16 rows x 64 ch -> two 16x32 fragments)
    v16bf qa[2];
    {
        const __bf16* qrow = Qp + (size_t)(qt + (lane & 15)) * CHN + (hi << 4);
        #pragma unroll
        for (int cc = 0; cc < 2; ++cc) {
            v8bf lo = *(const v8bf*)(qrow + cc * 32);
            v8bf hg = *(const v8bf*)(qrow + cc * 32 + 8);
            #pragma unroll
            for (int i = 0; i < 8; ++i) { qa[cc][i] = lo[i]; qa[cc][i + 8] = hg[i]; }
        }
    }

    const float NEG_INF = -__builtin_inff();
    float mi[8], li[8];
    #pragma unroll
    for (int r = 0; r < 8; ++r) { mi[r] = NEG_INF; li[r] = 0.0f; }
    v8f o0 = {}, o1 = {}, o2 = {}, o3 = {};
    v8f* o[4] = { &o0, &o1, &o2, &o3 };

    const int jend = q0b + 128;
    for (int j0 = 0; j0 < jend; j0 += 32) {
        __syncthreads();
        {   // stage K chunk [32x64] and V chunk transposed [64x32]
            int key = threadIdx.x >> 3;
            int c0 = (threadIdx.x & 7) * 8;
            *(v8bf*)&ldsK[key * CHN + c0] = *(const v8bf*)(Kp + (size_t)(j0 + key) * CHN + c0);
            v8bf vv = *(const v8bf*)(Vp + (size_t)(j0 + key) * CHN + c0);
            #pragma unroll
            for (int e = 0; e < 8; ++e) ldsVt[(c0 + e) * 32 + key] = vv[e];
        }
        __syncthreads();

        // Scores: two 16x16 tiles over a 32-key chunk, contraction over 64 ch
        v8f s0 = {}, s1 = {};
        #pragma unroll
        for (int cc = 0; cc < 2; ++cc) {
            int cb = cc * 32 + (hi << 4);
            v16bf bk0 = *(const v16bf*)&ldsK[(lane & 15) * CHN + cb];
            v16bf bk1 = *(const v16bf*)&ldsK[((lane & 15) + 16) * CHN + cb];
            s0 = __builtin_amdgcn_wmma_f32_16x16x32_bf16(false, qa[cc], false, bk0,
                                                         (short)0, s0, false, false);
            s1 = __builtin_amdgcn_wmma_f32_16x16x32_bf16(false, qa[cc], false, bk1,
                                                         (short)0, s1, false, false);
        }

        const int k0 = j0 + (lane & 15);
        const int k1 = k0 + 16;
        const bool pm0 = pad[b * SS + k0];
        const bool pm1 = pad[b * SS + k1];
        float* pbuf = &ldsP[wave][0];

        #pragma unroll
        for (int r = 0; r < 8; ++r) {
            int m = qt + r + (hi << 3);
            float v0 = (pm0 || k0 > m) ? NEG_INF : s0[r];
            float v1 = (pm1 || k1 > m) ? NEG_INF : s1[r];
            // row max across the 16 lanes of this half-wave
            float mx = fmaxf(v0, v1);
            #pragma unroll
            for (int off = 1; off < 16; off <<= 1)
                mx = fmaxf(mx, __shfl_xor(mx, off, 32));
            float newm = fmaxf(mi[r], mx);
            float corr, e0, e1;
            if (newm == NEG_INF) { corr = 1.0f; e0 = 0.0f; e1 = 0.0f; }
            else {
                corr = __expf(mi[r] - newm);
                e0 = __expf(v0 - newm);
                e1 = __expf(v1 - newm);
            }
            float rs = e0 + e1;
            #pragma unroll
            for (int off = 1; off < 16; off <<= 1)
                rs += __shfl_xor(rs, off, 32);
            li[r] = li[r] * corr + rs;
            mi[r] = newm;
            (*o[0])[r] *= corr; (*o[1])[r] *= corr;
            (*o[2])[r] *= corr; (*o[3])[r] *= corr;
            int prow = r + (hi << 3);
            pbuf[prow * 32 + (lane & 15)]      = e0;
            pbuf[prow * 32 + 16 + (lane & 15)] = e1;
        }
        asm volatile("s_wait_dscnt 0" ::: "memory");   // wave-private LDS RAW

        // Re-read P in A-fragment layout (16x32, K = key index), as bf16
        v16bf pa;
        {
            const float* pr = pbuf + (lane & 15) * 32 + (hi << 4);
            #pragma unroll
            for (int i = 0; i < 8; ++i) {
                pa[i]     = (__bf16)pr[i];
                pa[i + 8] = (__bf16)pr[i + 8];
            }
        }
        // O += P @ V_chunk : 4 column blocks of 16 channels
        #pragma unroll
        for (int f = 0; f < 4; ++f) {
            v16bf vb = *(const v16bf*)&ldsVt[(f * 16 + (lane & 15)) * 32 + (hi << 4)];
            *o[f] = __builtin_amdgcn_wmma_f32_16x16x32_bf16(false, pa, false, vb,
                                                            (short)0, *o[f], false, false);
        }
    }

    // Normalize and store merged-head output [B,S,D] bf16
    #pragma unroll
    for (int r = 0; r < 8; ++r) {
        float inv = (li[r] > 0.0f) ? 1.0f / li[r] : 0.0f;
        int s = qt + r + (hi << 3);
        size_t base = ((size_t)b * SS + s) * DD + h * CHN;
        #pragma unroll
        for (int f = 0; f < 4; ++f)
            Obf[base + f * 16 + (lane & 15)] = (__bf16)((*o[f])[r] * inv);
    }
}

// ---------------------------------------------------------------------------
// Output projection: out = O @ Wo^T + bo, fp32 result.
// ---------------------------------------------------------------------------
__global__ __launch_bounds__(256) void out_proj_kernel(
    const __bf16* __restrict__ Obf, const __bf16* __restrict__ Wob,
    const float* __restrict__ bo, float* __restrict__ out) {
    const int lane = threadIdx.x & 31;
    const int wave = threadIdx.x >> 5;
    const int tile = blockIdx.x * 8 + wave;
    const int m0 = (tile >> 5) * 32;
    const int n0 = (tile & 31) * 32;

    v8f acc[4] = { {}, {}, {}, {} };
    gemm_tile_2x2(Obf, Wob, m0, n0, DD, lane, acc);

    #pragma unroll
    for (int mt = 0; mt < 2; ++mt) {
        #pragma unroll
        for (int nt = 0; nt < 2; ++nt) {
            const v8f& a = acc[mt * 2 + nt];
            const int n = n0 + nt * 16 + (lane & 15);
            const float bi = bo[n];
            #pragma unroll
            for (int r = 0; r < 8; ++r) {
                int m = m0 + mt * 16 + r + ((lane >> 4) << 3);
                out[(size_t)m * DD + n] = a[r] + bi;
            }
        }
    }
}

// ---------------------------------------------------------------------------
extern "C" void kernel_launch(void* const* d_in, const int* in_sizes, int n_in,
                              void* d_out, int out_size, void* d_ws, size_t ws_size,
                              hipStream_t stream) {
    const float* x_q  = (const float*)d_in[0];
    const float* x_kv = (const float*)d_in[1];
    const bool*  pad  = (const bool*)d_in[2];
    const float* Wq = (const float*)d_in[3];
    const float* bq = (const float*)d_in[4];
    const float* Wk = (const float*)d_in[5];
    const float* bk = (const float*)d_in[6];
    const float* Wv = (const float*)d_in[7];
    const float* bv = (const float*)d_in[8];
    const float* Wo = (const float*)d_in[9];
    const float* bo = (const float*)d_in[10];
    float* out = (float*)d_out;

    char* w = (char*)d_ws;
    const size_t MB = 1u << 20;
    __bf16* xq_bf  = (__bf16*)(w + 0 * MB);   // 8 MiB
    __bf16* xkv_bf = (__bf16*)(w + 8 * MB);   // 8 MiB
    __bf16* Wq_bf  = (__bf16*)(w + 16 * MB);  // 2 MiB each
    __bf16* Wk_bf  = (__bf16*)(w + 18 * MB);
    __bf16* Wv_bf  = (__bf16*)(w + 20 * MB);
    __bf16* Wo_bf  = (__bf16*)(w + 22 * MB);
    __bf16* Qh     = (__bf16*)(w + 24 * MB);  // [B,H,S,CH] 8 MiB
    __bf16* Kh     = (__bf16*)(w + 32 * MB);
    __bf16* Vh     = (__bf16*)(w + 40 * MB);
    __bf16* Obf    = (__bf16*)(w + 48 * MB);  // [B,S,D]   8 MiB

    const int nX = BB * SS * DD;   // 4,194,304
    const int nW = DD * DD;        // 1,048,576

    cvt_bf16_kernel<<<(nX / 4) / 256, 256, 0, stream>>>(x_q,  xq_bf,  nX / 4);
    cvt_bf16_kernel<<<(nX / 4) / 256, 256, 0, stream>>>(x_kv, xkv_bf, nX / 4);
    cvt_bf16_kernel<<<(nW / 4) / 256, 256, 0, stream>>>(Wq, Wq_bf, nW / 4);
    cvt_bf16_kernel<<<(nW / 4) / 256, 256, 0, stream>>>(Wk, Wk_bf, nW / 4);
    cvt_bf16_kernel<<<(nW / 4) / 256, 256, 0, stream>>>(Wv, Wv_bf, nW / 4);
    cvt_bf16_kernel<<<(nW / 4) / 256, 256, 0, stream>>>(Wo, Wo_bf, nW / 4);

    proj_qkv_kernel<<<dim3(512, 1, 3), 256, 0, stream>>>(
        xq_bf, xkv_bf, Wq_bf, Wk_bf, Wv_bf, bq, bk, bv, Qh, Kh, Vh);

    attn_kernel<<<dim3(SS / 128, BB * HH), 256, 0, stream>>>(Qh, Kh, Vh, pad, Obf);

    out_proj_kernel<<<512, 256, 0, stream>>>(Obf, Wo_bf, bo, out);

    (void)in_sizes; (void)n_in; (void)out_size; (void)ws_size;
}
